// DifferentialDropout_13417477833110
// MI455X (gfx1250) — matrix operand order repaired and checked by reading, not compiled, not run on hardware
//
#include <hip/hip_runtime.h>
#include <hip/hip_bf16.h>
#include <math.h>

#define DD   16384        // elements per row
#define BB   1024         // rows
#define NBIN 1000
#define NTOT (16777216.0f)
#define NKSTEP (DD / 32)  // 512

typedef __attribute__((ext_vector_type(16))) __bf16 v16bf;
typedef __attribute__((ext_vector_type(8)))  __bf16 v8bf;
typedef __attribute__((ext_vector_type(8)))  float  v8f;

// ---- workspace float offsets (first 16MB region) ----
static constexpr size_t OFF_RS   = 0;        // row sums      [1024]
static constexpr size_t OFF_RSQ  = 1024;     // row sumsq     [1024]
static constexpr size_t OFF_RMIN = 2048;     // row min       [1024]
static constexpr size_t OFF_RMAX = 3072;     // row max       [1024]
static constexpr size_t OFF_SC   = 4096;     // scalars: gmean, inv_std, lo, width, inv_width
static constexpr size_t OFF_FACT = 4352;     // factor1       [1024]
static constexpr size_t OFF_LENT = 5376;     // local entropy [1024]
static constexpr size_t OFF_KEEP = 6400;     // keep          [1024]
static constexpr size_t OFF_P    = 7424;     // p             [1024]
static constexpr size_t OFF_INVD = 8448;     // 1/sqrt(diag)  [1024]
static constexpr size_t OFF_BENT = 9472;     // batch entropy [1]
static constexpr size_t OFF_BCNT = 9728;     // batch counts  [1000]
static constexpr size_t OFF_RCNT = 1048576;  // row counts    [1024*1000]
static constexpr size_t OFF_COV  = 2621440;  // cov           [1024*1024]
static constexpr size_t TC_BYTE_OFF  = (size_t)16 << 20;  // bf16 centered  [1024][16384]
static constexpr size_t TCT_BYTE_OFF = (size_t)48 << 20;  // bf16 transpose [16384][1024]

// -------------------- stage 1: per-row stats --------------------
__global__ __launch_bounds__(256) void k_row_stats(const float* __restrict__ x,
                                                   float* __restrict__ ws) {
  __shared__ float s0[256], s1[256], s2[256], s3[256];
  const int b = blockIdx.x, tid = threadIdx.x;
  const float* row = x + (size_t)b * DD;
  float s = 0.f, sq = 0.f, mn = 3.4e38f, mx = -3.4e38f;
  for (int i = tid; i < DD; i += 256) {
    float v = row[i];
    s += v; sq += v * v; mn = fminf(mn, v); mx = fmaxf(mx, v);
  }
  s0[tid] = s; s1[tid] = sq; s2[tid] = mn; s3[tid] = mx;
  __syncthreads();
  for (int st = 128; st > 0; st >>= 1) {
    if (tid < st) {
      s0[tid] += s0[tid + st];
      s1[tid] += s1[tid + st];
      s2[tid] = fminf(s2[tid], s2[tid + st]);
      s3[tid] = fmaxf(s3[tid], s3[tid + st]);
    }
    __syncthreads();
  }
  if (tid == 0) {
    ws[OFF_RS + b] = s0[0]; ws[OFF_RSQ + b] = s1[0];
    ws[OFF_RMIN + b] = s2[0]; ws[OFF_RMAX + b] = s3[0];
  }
}

// -------------------- stage 2: global scalars --------------------
__global__ __launch_bounds__(256) void k_global_scalars(float* __restrict__ ws) {
  __shared__ float s0[256], s1[256], s2[256], s3[256];
  const int tid = threadIdx.x;
  float s = 0.f, sq = 0.f, mn = 3.4e38f, mx = -3.4e38f;
  for (int i = tid; i < BB; i += 256) {
    s += ws[OFF_RS + i]; sq += ws[OFF_RSQ + i];
    mn = fminf(mn, ws[OFF_RMIN + i]); mx = fmaxf(mx, ws[OFF_RMAX + i]);
  }
  s0[tid] = s; s1[tid] = sq; s2[tid] = mn; s3[tid] = mx;
  __syncthreads();
  for (int st = 128; st > 0; st >>= 1) {
    if (tid < st) {
      s0[tid] += s0[tid + st];
      s1[tid] += s1[tid + st];
      s2[tid] = fminf(s2[tid], s2[tid + st]);
      s3[tid] = fmaxf(s3[tid], s3[tid + st]);
    }
    __syncthreads();
  }
  if (tid == 0) {
    float sum = s0[0], sumsq = s1[0];
    float gmean = sum / NTOT;
    float var = (sumsq - sum * sum / NTOT) / (NTOT - 1.0f);  // ddof=1
    float gstd = sqrtf(var);
    float inv_std = 1.0f / gstd;
    float lo = (s2[0] - gmean) * inv_std;
    float hi = (s3[0] - gmean) * inv_std;
    float width = (hi - lo) / (float)NBIN;
    ws[OFF_SC + 0] = gmean;
    ws[OFF_SC + 1] = inv_std;
    ws[OFF_SC + 2] = lo;
    ws[OFF_SC + 3] = width;
    ws[OFF_SC + 4] = 1.0f / width;
  }
}

// ---------- stage 3: center->bf16 + per-row LDS histogram ----------
__global__ __launch_bounds__(256) void k_center_hist(const float* __restrict__ x,
                                                     float* __restrict__ ws,
                                                     __bf16* __restrict__ tc) {
  __shared__ unsigned hist[NBIN];
  const int b = blockIdx.x, tid = threadIdx.x;
  for (int i = tid; i < NBIN; i += 256) hist[i] = 0u;
  __syncthreads();
  const float rmean   = ws[OFF_RS + b] * (1.0f / DD);
  const float gmean   = ws[OFF_SC + 0];
  const float inv_std = ws[OFF_SC + 1];
  const float lo      = ws[OFF_SC + 2];
  const float invw    = ws[OFF_SC + 4];
  const float* row = x + (size_t)b * DD;
  for (int i = tid; i < DD; i += 256) {
    float v = row[i];
    tc[(size_t)b * DD + i] = (__bf16)(v - rmean);
    float t = (v - gmean) * inv_std;
    int idx = (int)floorf((t - lo) * invw);
    idx = min(max(idx, 0), NBIN - 1);
    atomicAdd(&hist[idx], 1u);
  }
  __syncthreads();
  for (int i = tid; i < NBIN; i += 256)
    ws[OFF_RCNT + (size_t)b * NBIN + i] = (float)hist[i];
}

// -------------------- stage 4: bf16 transpose (LDS tiles) --------------------
__global__ __launch_bounds__(256) void k_transpose(const __bf16* __restrict__ tc,
                                                   __bf16* __restrict__ tct) {
  __shared__ __bf16 tile[64][65];
  const int k0 = blockIdx.x * 64, b0 = blockIdx.y * 64;
  const int tx = threadIdx.x & 63, ty = threadIdx.x >> 6;  // 0..3
#pragma unroll
  for (int i = 0; i < 64; i += 4)
    tile[ty + i][tx] = tc[(size_t)(b0 + ty + i) * DD + k0 + tx];
  __syncthreads();
#pragma unroll
  for (int i = 0; i < 64; i += 4)
    tct[(size_t)(k0 + ty + i) * BB + b0 + tx] = tile[tx][ty + i];
}

// ---------------- stage 5: WMMA bf16 GEMM cov = tc @ tc^T ----------------
// 8 waves/WG (4 M-waves x 2 N-waves); wave = 32x64; WG tile = 128x128; grid 8x8.
// Double-buffered LDS K-panels filled with global_load_async_to_lds_b128
// (ASYNCcnt), A fragments read from LDS with ds_load_b128, B per bf16 element, 8 WMMAs per K-step.
__global__ __launch_bounds__(256) void k_gemm_wmma(const __bf16* __restrict__ tc,
                                                   const __bf16* __restrict__ tct,
                                                   float* __restrict__ cov) {
  constexpr int A_STRIDE  = 80;                   // 64B row + 16B pad (bank spread)
  constexpr int B_STRIDE  = 272;                  // 256B row + 16B pad (bank spread)
  constexpr int A_BYTES   = 128 * A_STRIDE;       // 10240
  constexpr int B_BYTES   = 32 * B_STRIDE;        // 8704
  constexpr int BUF_BYTES = A_BYTES + B_BYTES;    // 18944
  __shared__ __align__(16) unsigned char smem[2][BUF_BYTES];

  const int tid  = threadIdx.x;
  const int lane = tid & 31;
  const int w    = tid >> 5;
  const int wm   = w & 3;        // M strip of 32
  const int wn   = w >> 2;       // N strip of 64
  const int half = lane >> 4;
  const int ml   = lane & 15;
  const int Mbase = blockIdx.y * 128;
  const int Nbase = blockIdx.x * 128;

  // Per-thread async-copy assignment: 2 A chunks + 2 B chunks (16B each) per K-step.
  // A panel: 128 rows x 32 K (row stride A_STRIDE); chunk c: row=c>>2, q=c&3.
  // B panel: 32 K-rows x 128 N (row stride B_STRIDE); chunk c: row=c>>4, q=c&15.
  const int ca0 = tid, ca1 = tid + 256;
  const unsigned a0_lds = (unsigned)((ca0 >> 2) * A_STRIDE + (ca0 & 3) * 16);
  const unsigned a1_lds = (unsigned)((ca1 >> 2) * A_STRIDE + (ca1 & 3) * 16);
  const unsigned b0_lds = (unsigned)(A_BYTES + (ca0 >> 4) * B_STRIDE + (ca0 & 15) * 16);
  const unsigned b1_lds = (unsigned)(A_BYTES + (ca1 >> 4) * B_STRIDE + (ca1 & 15) * 16);
  const __bf16* a0_g = tc  + (size_t)(Mbase + (ca0 >> 2)) * DD + (ca0 & 3) * 8;
  const __bf16* a1_g = tc  + (size_t)(Mbase + (ca1 >> 2)) * DD + (ca1 & 3) * 8;
  const __bf16* b0_g = tct + (size_t)(ca0 >> 4) * BB + Nbase + (ca0 & 15) * 8;
  const __bf16* b1_g = tct + (size_t)(ca1 >> 4) * BB + Nbase + (ca1 & 15) * 8;

  // LDS aperture puts the LDS byte offset in addr[31:0] of the flat pointer.
  const unsigned smem_off = (unsigned)(uintptr_t)(&smem[0][0]);

  auto issue_panel = [&](unsigned bufoff, int kb) {
    unsigned long long ga0 = (unsigned long long)(uintptr_t)(a0_g + (size_t)kb * 32);
    unsigned long long ga1 = (unsigned long long)(uintptr_t)(a1_g + (size_t)kb * 32);
    unsigned long long gb0 = (unsigned long long)(uintptr_t)(b0_g + (size_t)kb * 32 * BB);
    unsigned long long gb1 = (unsigned long long)(uintptr_t)(b1_g + (size_t)kb * 32 * BB);
    asm volatile("global_load_async_to_lds_b128 %0, %1, off"
                 :: "v"(smem_off + bufoff + a0_lds), "v"(ga0) : "memory");
    asm volatile("global_load_async_to_lds_b128 %0, %1, off"
                 :: "v"(smem_off + bufoff + a1_lds), "v"(ga1) : "memory");
    asm volatile("global_load_async_to_lds_b128 %0, %1, off"
                 :: "v"(smem_off + bufoff + b0_lds), "v"(gb0) : "memory");
    asm volatile("global_load_async_to_lds_b128 %0, %1, off"
                 :: "v"(smem_off + bufoff + b1_lds), "v"(gb1) : "memory");
  };

  v8f acc[2][4];
#pragma unroll
  for (int mt = 0; mt < 2; ++mt)
#pragma unroll
    for (int nt = 0; nt < 4; ++nt)
#pragma unroll
      for (int r = 0; r < 8; ++r) acc[mt][nt][r] = 0.0f;

  issue_panel(0u, 0);

  for (int kb = 0; kb < NKSTEP; ++kb) {
    const unsigned cur = (kb & 1) ? (unsigned)BUF_BYTES : 0u;
    if (kb + 1 < NKSTEP) {
      issue_panel(cur ^ (unsigned)BUF_BYTES, kb + 1);
      // 4 loads in flight for next buffer; async loads complete in order, so
      // ASYNCcnt<=4 means the current buffer's 4 copies have landed.
      asm volatile("s_wait_asynccnt 0x4" ::: "memory");
    } else {
      asm volatile("s_wait_asynccnt 0x0" ::: "memory");
    }
    __syncthreads();  // all waves' copies for `cur` visible

    const unsigned char* sbuf = &smem[0][0] + cur;

    // A fragments: lane holds row M=ml(+16*mt); K chunks {8h..8h+7}, {16+8h..23+8h}
    v16bf afrag[2];
#pragma unroll
    for (int mt = 0; mt < 2; ++mt) {
      const unsigned char* pa = sbuf + (wm * 32 + mt * 16 + ml) * A_STRIDE + 16 * half;
      v8bf x0 = *(const v8bf*)pa;
      v8bf x1 = *(const v8bf*)(pa + 32);
      afrag[mt] = __builtin_shufflevector(x0, x1, 0, 1, 2, 3, 4, 5, 6, 7,
                                                  8, 9, 10, 11, 12, 13, 14, 15);
    }
    // B fragments: lane holds column N=ml; K chunks {8h..8h+7}, {16+8h..23+8h}
    v16bf bfrag[4];
#pragma unroll
    for (int nt = 0; nt < 4; ++nt) {
      const unsigned char* pb = sbuf + A_BYTES + (wn * 64 + nt * 16 + ml) * 2;
#pragma unroll
      for (int s = 0; s < 16; ++s) {
        int kk = (s < 8) ? (8 * half + s) : (16 + 8 * half + (s - 8));
        bfrag[nt][s] = *(const __bf16*)(pb + kk * B_STRIDE);
      }
    }
#pragma unroll
    for (int mt = 0; mt < 2; ++mt)
#pragma unroll
      for (int nt = 0; nt < 4; ++nt)
        acc[mt][nt] = __builtin_amdgcn_wmma_f32_16x16x32_bf16(
            false, afrag[mt], false, bfrag[nt], (short)0, acc[mt][nt], false, false);

    __syncthreads();  // done reading `cur`; next iter may overwrite it
  }

  // C/D layout: VGPR r -> M = r + 8*half; N = ml
#pragma unroll
  for (int mt = 0; mt < 2; ++mt)
#pragma unroll
    for (int nt = 0; nt < 4; ++nt)
#pragma unroll
      for (int r = 0; r < 8; ++r) {
        int rowi = Mbase + wm * 32 + mt * 16 + 8 * half + r;
        int coli = Nbase + wn * 64 + nt * 16 + ml;
        cov[(size_t)rowi * BB + coli] = acc[mt][nt][r];
      }
}

// -------------------- stage 6: inv diag --------------------
__global__ __launch_bounds__(256) void k_invd(const float* __restrict__ cov,
                                              float* __restrict__ ws) {
  int i = blockIdx.x * 256 + threadIdx.x;
  if (i < BB) ws[OFF_INVD + i] = 1.0f / sqrtf(cov[(size_t)i * BB + i]);
}

// -------------------- stage 7: factor1 = mean |corr| per row --------------------
__global__ __launch_bounds__(256) void k_factor1(const float* __restrict__ cov,
                                                 float* __restrict__ ws) {
  __shared__ float sm[256];
  const int i = blockIdx.x, tid = threadIdx.x;
  float v = 0.f;
  for (int j = tid; j < BB; j += 256)
    v += fabsf(cov[(size_t)i * BB + j]) * ws[OFF_INVD + j];
  sm[tid] = v;
  __syncthreads();
  for (int st = 128; st > 0; st >>= 1) {
    if (tid < st) sm[tid] += sm[tid + st];
    __syncthreads();
  }
  if (tid == 0) ws[OFF_FACT + i] = sm[0] * ws[OFF_INVD + i] * (1.0f / BB);
}

// -------------------- stage 8: batch counts (sum rows per bin) --------------------
__global__ __launch_bounds__(256) void k_batch_counts(float* __restrict__ ws) {
  __shared__ float sm[256];
  const int bin = blockIdx.x, tid = threadIdx.x;
  float v = 0.f;
  for (int r = tid; r < BB; r += 256)
    v += ws[OFF_RCNT + (size_t)r * NBIN + bin];
  sm[tid] = v;
  __syncthreads();
  for (int st = 128; st > 0; st >>= 1) {
    if (tid < st) sm[tid] += sm[tid + st];
    __syncthreads();
  }
  if (tid == 0) ws[OFF_BCNT + bin] = sm[0];
}

// -------------------- stage 9: batch entropy --------------------
__global__ __launch_bounds__(256) void k_batch_entropy(float* __restrict__ ws) {
  __shared__ float sm[256];
  const int tid = threadIdx.x;
  const float inv_nw = 1.0f / (NTOT * ws[OFF_SC + 3]);
  float v = 0.f;
  for (int i = tid; i < NBIN; i += 256) {
    float c = ws[OFF_BCNT + i];
    if (c > 0.f) { float p = c * inv_nw; v -= p * log2f(p); }
  }
  sm[tid] = v;
  __syncthreads();
  for (int st = 128; st > 0; st >>= 1) {
    if (tid < st) sm[tid] += sm[tid + st];
    __syncthreads();
  }
  if (tid == 0) ws[OFF_BENT] = sm[0];
}

// -------------------- stage 10: local entropy per row --------------------
__global__ __launch_bounds__(256) void k_local_entropy(float* __restrict__ ws) {
  __shared__ float sm[256];
  const int row = blockIdx.x, tid = threadIdx.x;
  const float inv_nw = 1.0f / ((float)DD * ws[OFF_SC + 3]);
  float v = 0.f;
  for (int i = tid; i < NBIN; i += 256) {
    float c = ws[OFF_RCNT + (size_t)row * NBIN + i];
    if (c > 0.f) { float p = c * inv_nw; v -= p * log2f(p); }
  }
  sm[tid] = v;
  __syncthreads();
  for (int st = 128; st > 0; st >>= 1) {
    if (tid < st) sm[tid] += sm[tid + st];
    __syncthreads();
  }
  if (tid == 0) ws[OFF_LENT + row] = sm[0];
}

// -------------------- stage 11: keep / p per row --------------------
__global__ __launch_bounds__(256) void k_keep_p(float* __restrict__ ws) {
  int i = blockIdx.x * 256 + threadIdx.x;
  if (i >= BB) return;
  float be = ws[OFF_BENT];
  float f2 = ws[OFF_LENT + i] / be;
  float f1, F2;
  if (f2 == 0.0f) { f1 = 0.0f; F2 = 1.0f; }
  else            { f1 = ws[OFF_FACT + i]; F2 = (f2 < 1.0f) ? 1.0f / f2 : f2; }
  float keep = f1 / F2;
  ws[OFF_KEEP + i] = keep;
  ws[OFF_P + i] = 1.0f - keep;
}

// -------------------- stage 12: masked scaled output --------------------
__global__ __launch_bounds__(256) void k_output(const float* __restrict__ x,
                                                const float* __restrict__ u,
                                                const float* __restrict__ ws,
                                                float* __restrict__ out) {
  int i4 = blockIdx.x * 256 + threadIdx.x;         // 4 elements per thread
  size_t base = (size_t)i4 * 4;
  int b = (int)(base >> 14);                        // D = 16384 = 2^14
  float keep = ws[OFF_KEEP + b];
  float p = ws[OFF_P + b];
  float inv_keep = 1.0f / keep;                     // matches ref (NaN if keep==0)
  float4 xv = ((const float4*)x)[i4];
  float4 uv = ((const float4*)u)[i4];
  float4 o;
  o.x = ((uv.x > p) ? 1.0f : 0.0f) * xv.x * inv_keep;
  o.y = ((uv.y > p) ? 1.0f : 0.0f) * xv.y * inv_keep;
  o.z = ((uv.z > p) ? 1.0f : 0.0f) * xv.z * inv_keep;
  o.w = ((uv.w > p) ? 1.0f : 0.0f) * xv.w * inv_keep;
  ((float4*)out)[i4] = o;
}

extern "C" void kernel_launch(void* const* d_in, const int* in_sizes, int n_in,
                              void* d_out, int out_size, void* d_ws, size_t ws_size,
                              hipStream_t stream) {
  const float* x = (const float*)d_in[0];
  const float* u = (const float*)d_in[1];
  float* ws = (float*)d_ws;
  float* out = (float*)d_out;
  __bf16* tc  = (__bf16*)((char*)d_ws + TC_BYTE_OFF);
  __bf16* tct = (__bf16*)((char*)d_ws + TCT_BYTE_OFF);
  float* cov = ws + OFF_COV;

  k_row_stats<<<BB, 256, 0, stream>>>(x, ws);
  k_global_scalars<<<1, 256, 0, stream>>>(ws);
  k_center_hist<<<BB, 256, 0, stream>>>(x, ws, tc);
  k_transpose<<<dim3(DD / 64, BB / 64), 256, 0, stream>>>(tc, tct);
  k_gemm_wmma<<<dim3(8, 8), 256, 0, stream>>>(tc, tct, cov);
  k_invd<<<4, 256, 0, stream>>>(cov, ws);
  k_factor1<<<BB, 256, 0, stream>>>(cov, ws);
  k_batch_counts<<<NBIN, 256, 0, stream>>>(ws);
  k_batch_entropy<<<1, 256, 0, stream>>>(ws);
  k_local_entropy<<<BB, 256, 0, stream>>>(ws);
  k_keep_p<<<4, 256, 0, stream>>>(ws);
  k_output<<<(BB * DD) / (256 * 4), 256, 0, stream>>>(x, u, ws, out);
}
